// RNN_55860344652144
// MI455X (gfx1250) — matrix operand order repaired and checked by reading, not compiled, not run on hardware
//
#include <hip/hip_runtime.h>

// ---------------------------------------------------------------------------
// LSTM char-RNN forward on MI455X (gfx1250), WMMA f16 -> f32 accumulate.
// Per step:  z = E[char_t] + H @ W_h   (E has bias folded in)
//            c = c*sig(z_f+1) + sig(z_i)*tanh(z_j);  h = tanh(c)*sig(z_o)
// Final:     out = H_final @ dense_w + dense_b
//
// Step kernel: one wave owns a 32(batch) x 16(unit) tile with all 4 gates
// (8 f32 accumulators). B fragments (gate weight columns) are loaded once
// per K-chunk and reused for both A row-tiles -> 12 b128 loads per 8 WMMAs.
// ---------------------------------------------------------------------------

typedef _Float16 half_t;
typedef _Float16 v8h  __attribute__((ext_vector_type(8)));
typedef _Float16 v16h __attribute__((ext_vector_type(16)));
typedef float    v8f  __attribute__((ext_vector_type(8)));

#define BATCH   512
#define UNITS   1024
#define NCHARS  128
#define SEQ     256
#define GATES4  4096   // 4*UNITS

// ---------------------------------------------------------------------------
// Fast transcendentals: gfx1250 has native V_TANH_F32.
// ---------------------------------------------------------------------------

__device__ __forceinline__ float fast_tanh(float x) {
#if __has_builtin(__builtin_amdgcn_tanhf)
    return __builtin_amdgcn_tanhf(x);
#else
    return 2.0f / (1.0f + __expf(-2.0f * x)) - 1.0f;
#endif
}

__device__ __forceinline__ float sigm(float x) {
    return 0.5f + 0.5f * fast_tanh(0.5f * x);
}

__device__ __forceinline__ v16h cat8(v8h lo, v8h hi) {
    return __builtin_shufflevector(lo, hi, 0,1,2,3,4,5,6,7,8,9,10,11,12,13,14,15);
}

// ---------------------------------------------------------------------------
// Preprocessing kernels (one-time per launch)
// ---------------------------------------------------------------------------

// E[c][n] = kernel[c][n] + bias[n]   (input one-hot matmul == row gather)
__global__ void build_embed(const float* __restrict__ kern,
                            const float* __restrict__ bias,
                            float* __restrict__ E) {
    int idx = blockIdx.x * 256 + threadIdx.x;
    if (idx < NCHARS * GATES4) {
        int n = idx % GATES4;
        E[idx] = kern[idx] + bias[n];
    }
}

// WhT[n][k] = kernel[128+k][n]  as f16 (column-major B for contiguous K loads)
__global__ void build_whT(const float* __restrict__ kern,
                          half_t* __restrict__ WhT) {
    int idx = blockIdx.x * 256 + threadIdx.x;
    if (idx < GATES4 * UNITS) {
        int n = idx / UNITS;
        int k = idx % UNITS;
        WhT[idx] = (half_t)kern[(NCHARS + k) * GATES4 + n];
    }
}

// dwT[n][k] = dense_w[k][n] as f16
__global__ void build_dwT(const float* __restrict__ dw,
                          half_t* __restrict__ dwT) {
    int idx = blockIdx.x * 256 + threadIdx.x;
    if (idx < NCHARS * UNITS) {
        int n = idx / UNITS;
        int k = idx % UNITS;
        dwT[idx] = (half_t)dw[k * NCHARS + n];
    }
}

// zero h0 (f16) and c (f32)
__global__ void zero_state(half_t* __restrict__ H0, float* __restrict__ C) {
    int idx = blockIdx.x * 256 + threadIdx.x;
    if (idx < BATCH * UNITS) {
        H0[idx] = (half_t)0.0f;
        C[idx]  = 0.0f;
    }
}

// ---------------------------------------------------------------------------
// LSTM step: wave = 32(batch) x 16(unit) x 4 gates.
// grid = (16, 16), block = 128 (4 waves; wave w -> unit tile blockIdx.x*4+w)
// ---------------------------------------------------------------------------

__global__ void __launch_bounds__(128)
lstm_step(const half_t* __restrict__ Hin,
          half_t* __restrict__ Hout,
          float* __restrict__ C,
          const float* __restrict__ E,
          const half_t* __restrict__ WhT,
          const int* __restrict__ chars,
          int t) {
    const int lane    = threadIdx.x & 31;
    const int wave    = threadIdx.x >> 5;
    const int nu      = blockIdx.x * 4 + wave;   // unit tile        [0,64)
    const int mb      = blockIdx.y;              // batch-32 group   [0,16)
    const int halfsel = lane >> 4;               // 0: lanes 0-15, 1: 16-31
    const int l16     = lane & 15;

    // A fragments: row-major Hin.
    // lanes 0-15 hold K{0-7,16-23}, lanes 16-31 hold K{8-15,24-31}.
    const half_t* Arow0 = Hin + (size_t)(mb * 32 + l16) * UNITS;
    const half_t* Arow1 = Arow0 + (size_t)16 * UNITS;
    const int     aoff  = halfsel * 8;

    // B fragments: WhT row-major [4096][1024] (K contiguous per column).
    // lanes 0-15 hold K{0-7,16-23}, lanes 16-31 hold K{8-15,24-31} of column l16.
    const half_t* Bcol[4];
#pragma unroll
    for (int g = 0; g < 4; ++g)
        Bcol[g] = WhT + (size_t)(g * UNITS + nu * 16 + l16) * UNITS;
    const int boff = halfsel * 8;

    v8f acc[2][4] = {};   // [row-tile][gate i,j,f,o], f32 accumulate

    for (int k = 0; k < UNITS; k += 32) {
        v16h a0 = cat8(*(const v8h*)(Arow0 + k + aoff),
                       *(const v8h*)(Arow0 + k + 16 + aoff));
        v16h a1 = cat8(*(const v8h*)(Arow1 + k + aoff),
                       *(const v8h*)(Arow1 + k + 16 + aoff));
#pragma unroll
        for (int g = 0; g < 4; ++g) {
            v16h b = cat8(*(const v8h*)(Bcol[g] + k + boff),
                          *(const v8h*)(Bcol[g] + k + 16 + boff));
            acc[0][g] = __builtin_amdgcn_wmma_f32_16x16x32_f16(
                false, a0, false, b, (short)0, acc[0][g], false, false);
            acc[1][g] = __builtin_amdgcn_wmma_f32_16x16x32_f16(
                false, a1, false, b, (short)0, acc[1][g], false, false);
        }
    }

    // Epilogue: C/D layout -> lane<16: M=r, lane>=16: M=r+8; N = l16.
    const int col = nu * 16 + l16;
#pragma unroll
    for (int m = 0; m < 2; ++m) {
#pragma unroll
        for (int r = 0; r < 8; ++r) {
            int row = mb * 32 + m * 16 + r + halfsel * 8;
            int ch  = chars[row * SEQ + t];
            const float* Er = E + (size_t)ch * GATES4 + col;
            float zi = acc[m][0][r] + Er[0 * UNITS];
            float zj = acc[m][1][r] + Er[1 * UNITS];
            float zf = acc[m][2][r] + Er[2 * UNITS];
            float zo = acc[m][3][r] + Er[3 * UNITS];

            size_t idx = (size_t)row * UNITS + col;
            float cprev = C[idx];
            float cnew  = cprev * sigm(zf + 1.0f) + sigm(zi) * fast_tanh(zj);
            float h     = fast_tanh(cnew) * sigm(zo);
            C[idx]    = cnew;
            Hout[idx] = (half_t)h;
        }
    }
}

// ---------------------------------------------------------------------------
// Final dense: out[512][128] = H @ dwT + db.  grid=(2,32), block=128.
// ---------------------------------------------------------------------------

__global__ void __launch_bounds__(128)
dense_out(const half_t* __restrict__ Hin,
          const half_t* __restrict__ dwT,
          const float* __restrict__ db,
          float* __restrict__ out) {
    const int lane    = threadIdx.x & 31;
    const int wave    = threadIdx.x >> 5;
    const int nu      = blockIdx.x * 4 + wave;   // col tile [0,8)
    const int mb      = blockIdx.y;              // row tile [0,32)
    const int halfsel = lane >> 4;
    const int l16     = lane & 15;

    const half_t* Arow = Hin + (size_t)(mb * 16 + l16) * UNITS;
    const int     aoff = halfsel * 8;
    const half_t* Bcol = dwT + (size_t)(nu * 16 + l16) * UNITS;
    const int     boff = halfsel * 8;

    v8f acc = {};
    for (int k = 0; k < UNITS; k += 32) {
        acc = __builtin_amdgcn_wmma_f32_16x16x32_f16(
            false,
            cat8(*(const v8h*)(Arow + k + aoff),
                 *(const v8h*)(Arow + k + 16 + aoff)),
            false,
            cat8(*(const v8h*)(Bcol + k + boff),
                 *(const v8h*)(Bcol + k + 16 + boff)),
            (short)0, acc, false, false);
    }

    const int col = nu * 16 + l16;
    float bcol = db[col];
#pragma unroll
    for (int r = 0; r < 8; ++r) {
        int row = mb * 16 + r + halfsel * 8;
        out[row * NCHARS + col] = acc[r] + bcol;
    }
}

// ---------------------------------------------------------------------------
// Launch
// ---------------------------------------------------------------------------

extern "C" void kernel_launch(void* const* d_in, const int* in_sizes, int n_in,
                              void* d_out, int out_size, void* d_ws, size_t ws_size,
                              hipStream_t stream) {
    const int*   chars = (const int*)d_in[0];    // [512, 256] char ids
    const float* kern  = (const float*)d_in[1];  // [1152, 4096]
    const float* bias  = (const float*)d_in[2];  // [4096]
    const float* dw    = (const float*)d_in[3];  // [1024, 128]
    const float* db    = (const float*)d_in[4];  // [128]
    float*       out   = (float*)d_out;          // [512, 128]

    char* ws = (char*)d_ws;
    size_t off = 0;
    auto carve = [&](size_t bytes) -> void* {
        void* p = ws + off;
        off = (off + bytes + 255) & ~(size_t)255;
        return p;
    };

    float*  E   = (float*) carve((size_t)NCHARS * GATES4 * 4);  // 2 MB
    half_t* WhT = (half_t*)carve((size_t)GATES4 * UNITS * 2);   // 8 MB
    half_t* dwT = (half_t*)carve((size_t)NCHARS * UNITS * 2);   // 256 KB
    half_t* H0  = (half_t*)carve((size_t)BATCH * UNITS * 2);    // 1 MB
    half_t* H1  = (half_t*)carve((size_t)BATCH * UNITS * 2);    // 1 MB
    float*  C   = (float*) carve((size_t)BATCH * UNITS * 4);    // 2 MB

    // --- preprocessing ---
    build_embed<<<(NCHARS * GATES4 + 255) / 256, 256, 0, stream>>>(kern, bias, E);
    build_whT  <<<(GATES4 * UNITS + 255) / 256, 256, 0, stream>>>(kern, WhT);
    build_dwT  <<<(NCHARS * UNITS + 255) / 256, 256, 0, stream>>>(dw, dwT);
    zero_state <<<(BATCH * UNITS + 255) / 256, 256, 0, stream>>>(H0, C);

    // --- 256 recurrent steps, H double-buffered ---
    half_t* Hbuf[2] = {H0, H1};
    dim3 sgrid(16, 16);   // 64 unit tiles / 4 waves, 16 batch-32 groups
    for (int t = 0; t < SEQ; ++t) {
        lstm_step<<<sgrid, 128, 0, stream>>>(
            Hbuf[t & 1], Hbuf[(t + 1) & 1], C, E, WhT, chars, t);
    }

    // --- final projection (h_final lives in Hbuf[SEQ & 1] == H0) ---
    dim3 dgrid(2, 32);    // 8 col tiles / 4 waves, 32 row tiles
    dense_out<<<dgrid, 128, 0, stream>>>(Hbuf[SEQ & 1], dwT, db, out);
}
